// W8A16Linear_24996709663383
// MI455X (gfx1250) — compile-verified, matrix operand order checked by reading, not run on hardware
//
#include <hip/hip_runtime.h>

typedef _Float16 v8h  __attribute__((ext_vector_type(8)));
typedef _Float16 v16h __attribute__((ext_vector_type(16)));
typedef float    v8f  __attribute__((ext_vector_type(8)));
typedef int      v4i  __attribute__((ext_vector_type(4)));

// ---- CDNA5 async global->LDS path (guarded; falls back to register staging)
#if defined(__AMDGCN__) && __has_builtin(__builtin_amdgcn_global_load_async_to_lds_b128)
#define USE_ASYNC_LDS 1
#else
#define USE_ASYNC_LDS 0
#endif

#if defined(__AMDGCN__) && __has_builtin(__builtin_amdgcn_s_wait_asynccnt)
#define WAIT_ASYNC(n) __builtin_amdgcn_s_wait_asynccnt(n)
#elif defined(__AMDGCN__)
#define WAIT_ASYNC(n) asm volatile("s_wait_asynccnt %0" ::"n"(n) : "memory")
#else
#define WAIT_ASYNC(n)
#endif

// Builtin signature (from hipcc diagnostic): arg0 = v4i in AS1 (global),
// arg1 = v4i in AS3 (LDS), arg2/arg3 = imm offset / cpol.
typedef __attribute__((address_space(1))) v4i* gv4i_p;
typedef __attribute__((address_space(3))) v4i* lv4i_p;

namespace {
constexpr int IN_F   = 4096;
constexpr int OUT_F  = 11008;
constexpr int M_TOT  = 4 * 2048;         // B*S = 8192
constexpr int MB     = 256;              // block tile M
constexpr int NB     = 128;              // block tile N
constexpr int KB     = 32;               // K step per WMMA round
constexpr int LDSW   = 40;               // LDS row stride in halfs (80B, conflict-free)
constexpr int KSTEPS = IN_F / KB;        // 128
}

union Frag { v16h v; v8h h[2]; };

// Issue one A-row slice (64B = 4 x b128) as async global->LDS copies.
__device__ __forceinline__ void issue_async_row(const _Float16* g, unsigned lbase) {
#if USE_ASYNC_LDS
#pragma unroll
    for (int c = 0; c < 4; ++c)
        __builtin_amdgcn_global_load_async_to_lds_b128(
            (gv4i_p)(unsigned long long)(g + c * 8),
            (lv4i_p)(unsigned)(lbase + c * 16), 0, 0);
#else
    (void)g; (void)lbase;
#endif
}

// D = A(16x32 f16) x B(32x16 f16) + C(f32); per-lane fragment layout per CDNA5 ISA:
// lane: idx = lane%16 (row for A / col for B), half = lane/16
// v16h elements 0..7  = K(half*8 .. half*8+7), 8..15 = K(16+half*8 .. +7)
__global__ __launch_bounds__(256)
void w8a16_wmma_kernel(const _Float16* __restrict__ X,
                       const int*      __restrict__ W8,
                       const _Float16* __restrict__ SC,
                       const _Float16* __restrict__ BI,
                       _Float16*       __restrict__ OUT)
{
    __shared__ _Float16 ldsA[2][MB * LDSW];  // activations, double-buffered (async)
    __shared__ _Float16 ldsB[NB * LDSW];     // weights as raw (f16)int8

    const int tid  = threadIdx.x;
    const int lane = tid & 31;
    const int wv   = tid >> 5;          // 0..7
    const int idx  = lane & 15;
    const int hlf  = lane >> 4;

    const int bm = blockIdx.y * MB;
    const int bn = blockIdx.x * NB;

    // 8 waves arranged 4(M) x 2(N); each wave computes 64x64 = 4x4 fragments
    const int wm = (wv & 3) << 6;       // 0,64,128,192
    const int wn = (wv >> 2) << 6;      // 0,64

    // A staging: thread owns one full 32-half row slice (4 x 16B) per K-step
    const _Float16* gA = X + (size_t)(bm + tid) * IN_F;
    const unsigned  lA = (unsigned)(unsigned long long)&ldsA[0][tid * LDSW];
    constexpr unsigned ABUF = MB * LDSW * sizeof(_Float16);   // bytes per A buffer

    // W staging: thread owns 16 ints (half a row) per K-step
    const int srow = tid >> 1;
    const int skh  = (tid & 1) << 4;
    const int* gW  = W8 + (size_t)(bn + srow) * IN_F + skh;
    _Float16*  sB  = ldsB + srow * LDSW + skh;

    v8f acc[4][4];
    const v8f vzero = {0.f, 0.f, 0.f, 0.f, 0.f, 0.f, 0.f, 0.f};
#pragma unroll
    for (int i = 0; i < 4; ++i)
#pragma unroll
        for (int j = 0; j < 4; ++j)
            acc[i][j] = vzero;

    // ---- prefetch K-step 0
#if USE_ASYNC_LDS
    issue_async_row(gA, lA);             // A tile 0 -> ldsA[0]
#else
    v8h ra[4];
#pragma unroll
    for (int c = 0; c < 4; ++c) ra[c] = *(const v8h*)(gA + c * 8);
#endif
    v4i rw0 = *(const v4i*)gW;
    v4i rw1 = *(const v4i*)(gW + 4);
    v4i rw2 = *(const v4i*)(gW + 8);
    v4i rw3 = *(const v4i*)(gW + 12);

    for (int kt = 0; kt < KSTEPS; ++kt) {
        __syncthreads();                 // previous round done reading LDS

        // stage W tile kt: int8 (in int32) -> f16 via i16 convert, store 32B
        v8h wh0, wh1;
#pragma unroll
        for (int e = 0; e < 4; ++e) {
            wh0[e]     = (_Float16)(short)rw0[e];
            wh0[e + 4] = (_Float16)(short)rw1[e];
            wh1[e]     = (_Float16)(short)rw2[e];
            wh1[e + 4] = (_Float16)(short)rw3[e];
        }
        *(v8h*)sB       = wh0;
        *(v8h*)(sB + 8) = wh1;

#if !USE_ASYNC_LDS
        {   // synchronous A staging (fallback)
            _Float16* sA = &ldsA[kt & 1][tid * LDSW];
#pragma unroll
            for (int c = 0; c < 4; ++c) *(v8h*)(sA + c * 8) = ra[c];
        }
#endif

        if (kt + 1 < KSTEPS) {
            gA += KB;
            gW += KB;
#if USE_ASYNC_LDS
            // issue A tile kt+1 into the other buffer, then drain tile kt:
            // in-order ASYNCcnt => waiting <=4 completes the 4 older requests.
            issue_async_row(gA, lA + ((kt + 1) & 1) * ABUF);
            WAIT_ASYNC(4);
#else
#pragma unroll
            for (int c = 0; c < 4; ++c) ra[c] = *(const v8h*)(gA + c * 8);
#endif
            rw0 = *(const v4i*)gW;
            rw1 = *(const v4i*)(gW + 4);
            rw2 = *(const v4i*)(gW + 8);
            rw3 = *(const v4i*)(gW + 12);
        }
#if USE_ASYNC_LDS
        else {
            WAIT_ASYNC(0);               // drain final A tile
        }
#endif

        __syncthreads();                 // A+W tiles of step kt visible

        // fragment loads from LDS (row stride 80B: conflict-free b128 reads)
        const _Float16* tA = ldsA[kt & 1];
        Frag a[4], b[4];
#pragma unroll
        for (int i = 0; i < 4; ++i) {
            const _Float16* p = tA + (wm + i * 16 + idx) * LDSW + hlf * 8;
            a[i].h[0] = *(const v8h*)p;
            a[i].h[1] = *(const v8h*)(p + 16);
        }
#pragma unroll
        for (int j = 0; j < 4; ++j) {
            const _Float16* p = ldsB + (wn + j * 16 + idx) * LDSW + hlf * 8;
            b[j].h[0] = *(const v8h*)p;
            b[j].h[1] = *(const v8h*)(p + 16);
        }

        // 16 WMMAs per K-step per wave
#pragma unroll
        for (int i = 0; i < 4; ++i)
#pragma unroll
            for (int j = 0; j < 4; ++j)
                acc[i][j] = __builtin_amdgcn_wmma_f32_16x16x32_f16(
                    false, a[i].v, false, b[j].v,
                    (short)0, acc[i][j], false, false);
    }

    // epilogue: C = scale[n]*acc + bias[n], store fp16.
    // C/D layout: VGPR r holds (M = r + 8*half, N = idx) of each 16x16 tile.
#pragma unroll
    for (int j = 0; j < 4; ++j) {
        const int n    = bn + wn + j * 16 + idx;
        const float s  = (float)SC[n];
        const float bb = (float)BI[n];
#pragma unroll
        for (int i = 0; i < 4; ++i) {
            const int m0 = bm + wm + i * 16 + hlf * 8;
#pragma unroll
            for (int r = 0; r < 8; ++r) {
                const float v = acc[i][j][r] * s + bb;
                OUT[(size_t)(m0 + r) * OUT_F + n] = (_Float16)v;
            }
        }
    }
}

extern "C" void kernel_launch(void* const* d_in, const int* in_sizes, int n_in,
                              void* d_out, int out_size, void* d_ws, size_t ws_size,
                              hipStream_t stream) {
    (void)in_sizes; (void)n_in; (void)out_size; (void)d_ws; (void)ws_size;
    const _Float16* x   = (const _Float16*)d_in[0];
    const int*      w8  = (const int*)d_in[1];
    const _Float16* sc  = (const _Float16*)d_in[2];
    const _Float16* bi  = (const _Float16*)d_in[3];
    _Float16*       out = (_Float16*)d_out;

    dim3 grid(OUT_F / NB, M_TOT / MB);   // (86, 32)
    w8a16_wmma_kernel<<<grid, 256, 0, stream>>>(x, w8, sc, bi, out);
}
